// AttentionModel_89644557402804
// MI455X (gfx1250) — matrix operand order repaired and checked by reading, not compiled, not run on hardware
//
#include <hip/hip_runtime.h>
#include <hip/hip_bf16.h>

// ---------------------------------------------------------------------------
// MI455X (gfx1250, wave32) fused attention block, f16 WMMA + f32 accumulate.
// ---------------------------------------------------------------------------

typedef __attribute__((ext_vector_type(16))) _Float16 v16h;
typedef __attribute__((ext_vector_type(8)))  float    v8f;

#define DIM      768
#define TDIM     2304
#define HEADS    12
#define HD       64
#define NSEQ     1024
#define BATCH    8
#define MROWS    (BATCH * NSEQ)          // 8192
#define SCALE    0.125f                  // 64^-0.5

// ---- WMMA wrapper (codegen-confirmed signature) ---------------------------
__device__ __forceinline__ v8f wmma16(v16h a, v16h b, v8f c) {
  return __builtin_amdgcn_wmma_f32_16x16x32_f16(
      false, a, false, b, (short)0, c, false, false);
}

// ---- A-fragment: 16x32 f16; lane holds row m=lane&15, K chunks kb..kb+7
// and kb+16..kb+23 with kb = 8*(lane>>4). Two 16B loads. ---------------------
__device__ __forceinline__ v16h load_a(const _Float16* base, int ld, int koff) {
  const int lane = threadIdx.x & 31;
  const _Float16* p = base + (size_t)(lane & 15) * ld + koff + ((lane >> 4) << 3);
  v16h r;
  reinterpret_cast<float4*>(&r)[0] = reinterpret_cast<const float4*>(p)[0];
  reinterpret_cast<float4*>(&r)[1] = reinterpret_cast<const float4*>(p + 16)[0];
  return r;
}

// ---- B-fragment: 32x16 f16 from column-major (B^T row-major) storage;
// lane holds column n=lane&15, K chunks kb..kb+7 and kb+16..kb+23, kb = 8*(lane>>4). --
__device__ __forceinline__ v16h load_b(const _Float16* baseT, int ld, int koff) {
  const int lane = threadIdx.x & 31;
  const _Float16* p = baseT + (size_t)(lane & 15) * ld + koff + ((lane >> 4) << 3);
  v16h r;
  reinterpret_cast<float4*>(&r)[0] = reinterpret_cast<const float4*>(p)[0];
  reinterpret_cast<float4*>(&r)[1] = reinterpret_cast<const float4*>(p + 16)[0];
  return r;
}

// ---- row reductions over the 16-lane half holding one C-fragment row ------
__device__ __forceinline__ float row_max16(float x) {
  #pragma unroll
  for (int m = 1; m <= 8; m <<= 1) x = fmaxf(x, __shfl_xor(x, m, 32));
  return x;
}
__device__ __forceinline__ float row_sum16(float x) {
  #pragma unroll
  for (int m = 1; m <= 8; m <<= 1) x += __shfl_xor(x, m, 32);
  return x;
}

// ===========================================================================
// Stage 0: precision conversion / weight transposition
// ===========================================================================
__global__ void cvt_f32_f16(const float* __restrict__ src,
                            _Float16* __restrict__ dst, int n) {
  int i = blockIdx.x * 256 + threadIdx.x;
  if (i < n) dst[i] = (_Float16)src[i];
}

// wt[c*rows + k] = w[k*cols + c]  (w is [rows=K, cols], wt is [cols, rows])
__global__ void trans_f32_f16(const float* __restrict__ w,
                              _Float16* __restrict__ wt, int rows, int cols) {
  int i = blockIdx.x * 256 + threadIdx.x;
  if (i < rows * cols) {
    int c = i / rows, k = i - c * rows;
    wt[i] = (_Float16)w[(size_t)k * cols + c];
  }
}

// ===========================================================================
// Stage 1: QKV GEMM  [8192,768] x [768,2304] + bias, scatter to q/k/vT f16
//   q,k: [B,H,N,64]   vT: [B,H,64,N]   (q pre-scaled by SCALE)
// One wave = 32 rows x 64 cols (B fragments reused by 2 row blocks);
// 8 accumulators, 24 K-steps, 8 independent WMMA per step.
// ===========================================================================
__global__ void __launch_bounds__(128)
qkv_gemm(const _Float16* __restrict__ xh,   // [8192,768]
         const _Float16* __restrict__ wT,   // [2304,768]
         const float*    __restrict__ bias, // [2304]
         _Float16* __restrict__ qb,
         _Float16* __restrict__ kbuf,
         _Float16* __restrict__ vtb) {
  const int lane = threadIdx.x & 31;
  const int task = blockIdx.x * 4 + (threadIdx.x >> 5);   // 9216 tasks
  const int rb = task / 36;                               // 256 row blocks (x32)
  const int cs = task - rb * 36;                          // 36 col strips (x64)
  const _Float16* a  = xh + (size_t)rb * 32 * DIM;
  const _Float16* b0 = wT + (size_t)cs * 64 * DIM;

  v8f acc[8] = {};
  for (int k0 = 0; k0 < DIM; k0 += 32) {
    __builtin_prefetch(a + (size_t)(lane & 15) * DIM + k0 + 64, 0, 3);
    v16h a0 = load_a(a,            DIM, k0);
    v16h a1 = load_a(a + 16 * DIM, DIM, k0);
    #pragma unroll
    for (int t = 0; t < 4; ++t) {
      v16h bf = load_b(b0 + t * 16 * DIM, DIM, k0);
      acc[t]     = wmma16(a0, bf, acc[t]);
      acc[4 + t] = wmma16(a1, bf, acc[4 + t]);
    }
  }

  const int half = lane >> 4, l15 = lane & 15;
  #pragma unroll
  for (int t = 0; t < 4; ++t) {
    const int c = cs * 64 + t * 16 + l15;       // global output column
    const int which = c / DIM;                  // 0=q 1=k 2=v
    const int rem = c - which * DIM;
    const int h = rem >> 6, d = rem & 63;
    const float bv = bias[c];
    #pragma unroll
    for (int rblk = 0; rblk < 2; ++rblk) {
      #pragma unroll
      for (int v = 0; v < 8; ++v) {
        const int r = rb * 32 + rblk * 16 + v + 8 * half;   // global row
        const int bi = r >> 10, n = r & 1023;
        const float val = acc[rblk * 4 + t][v] + bv;
        const size_t bh = (size_t)(bi * HEADS + h);
        if (which == 0)      qb  [(bh * NSEQ + n) * HD + d] = (_Float16)(val * SCALE);
        else if (which == 1) kbuf[(bh * NSEQ + n) * HD + d] = (_Float16)val;
        else                 vtb [(bh * HD + d) * NSEQ + n] = (_Float16)val;
      }
    }
  }
}

// ===========================================================================
// Stage 2: flash attention, one wave = 16 query rows, stream 32 keys/iter.
//   S = q kT (K=64, 4 WMMA) -> online softmax -> P v (K=32, 4 WMMA)
// ===========================================================================
__global__ void __launch_bounds__(128)
attn_kernel(const _Float16* __restrict__ qb,
            const _Float16* __restrict__ kbuf,
            const _Float16* __restrict__ vtb,
            _Float16* __restrict__ wa) {          // [8192,768]
  __shared__ __align__(32) _Float16 ptile[4][16 * 32];   // per-wave P staging
  const int lane = threadIdx.x & 31;
  const int wv = threadIdx.x >> 5;
  const int task = blockIdx.x * 4 + wv;           // 6144 tasks
  const int bh = task >> 6;                       // (b*12+h), 96
  const int qblk = task & 63;

  const _Float16* q  = qb   + ((size_t)bh * NSEQ + qblk * 16) * HD;
  const _Float16* kk = kbuf + (size_t)bh * NSEQ * HD;
  const _Float16* vt = vtb  + (size_t)bh * HD * NSEQ;

  const v16h aq0 = load_a(q, HD, 0);
  const v16h aq1 = load_a(q, HD, 32);

  float mi[8], li[8];
  v8f o0 = {}, o1 = {}, o2 = {}, o3 = {};
  #pragma unroll
  for (int v = 0; v < 8; ++v) { mi[v] = -1e30f; li[v] = 0.f; }

  const int half = lane >> 4, l15 = lane & 15;
  _Float16* lp = &ptile[wv][0];

  for (int j = 0; j < NSEQ; j += 32) {
    const _Float16* kp = kk + (size_t)j * HD;
    v8f s0 = {}, s1 = {};
    s0 = wmma16(aq0, load_b(kp,            HD, 0),  s0);
    s0 = wmma16(aq1, load_b(kp,            HD, 32), s0);
    s1 = wmma16(aq0, load_b(kp + 16 * HD,  HD, 0),  s1);
    s1 = wmma16(aq1, load_b(kp + 16 * HD,  HD, 32), s1);

    #pragma unroll
    for (int v = 0; v < 8; ++v) {
      const float rm = row_max16(fmaxf(s0[v], s1[v]));
      const float mn = fmaxf(mi[v], rm);
      const float alpha = __expf(mi[v] - mn);
      const float p0 = __expf(s0[v] - mn);
      const float p1 = __expf(s1[v] - mn);
      li[v] = li[v] * alpha + row_sum16(p0 + p1);
      mi[v] = mn;
      s0[v] = p0; s1[v] = p1;
      o0[v] *= alpha; o1[v] *= alpha; o2[v] *= alpha; o3[v] *= alpha;
      // C-frag -> LDS (row m = v+8*half, ld = 32)
      lp[(v + 8 * half) * 32 + l15]      = (_Float16)p0;
      lp[(v + 8 * half) * 32 + 16 + l15] = (_Float16)p1;
    }
    // LDS -> A-fragment (same-wave DS ops are in order; private tile)
    const v16h pa = load_a(lp, 32, 0);
    o0 = wmma16(pa, load_b(vt + 0 * 16 * NSEQ, NSEQ, j), o0);
    o1 = wmma16(pa, load_b(vt + 1 * 16 * NSEQ, NSEQ, j), o1);
    o2 = wmma16(pa, load_b(vt + 2 * 16 * NSEQ, NSEQ, j), o2);
    o3 = wmma16(pa, load_b(vt + 3 * 16 * NSEQ, NSEQ, j), o3);
  }

  const int b = bh / HEADS, h = bh - b * HEADS;
  #pragma unroll
  for (int v = 0; v < 8; ++v) {
    const int m = v + 8 * half;
    const size_t row = (size_t)b * NSEQ + qblk * 16 + m;
    const float inv = 1.0f / li[v];
    _Float16* wr = wa + row * DIM + h * HD + l15;
    wr[0]  = (_Float16)(o0[v] * inv);
    wr[16] = (_Float16)(o1[v] * inv);
    wr[32] = (_Float16)(o2[v] * inv);
    wr[48] = (_Float16)(o3[v] * inv);
  }
}

// ===========================================================================
// Stage 3: projection GEMM  [8192,768] x [768,768] + bias -> fp32 out
// One wave = 32 rows x 64 cols, same blocking as Stage 1.
// ===========================================================================
__global__ void __launch_bounds__(128)
proj_gemm(const _Float16* __restrict__ wa,   // [8192,768]
          const _Float16* __restrict__ wT,   // [768,768] (transposed)
          const float*    __restrict__ bias, // [768]
          float* __restrict__ out) {         // [8192,768]
  const int lane = threadIdx.x & 31;
  const int task = blockIdx.x * 4 + (threadIdx.x >> 5);  // 3072 tasks
  const int rb = task / 12;                              // 256 row blocks (x32)
  const int cs = task - rb * 12;                         // 12 col strips (x64)
  const _Float16* a  = wa + (size_t)rb * 32 * DIM;
  const _Float16* b0 = wT + (size_t)cs * 64 * DIM;

  v8f acc[8] = {};
  for (int k0 = 0; k0 < DIM; k0 += 32) {
    __builtin_prefetch(a + (size_t)(lane & 15) * DIM + k0 + 64, 0, 3);
    v16h a0 = load_a(a,            DIM, k0);
    v16h a1 = load_a(a + 16 * DIM, DIM, k0);
    #pragma unroll
    for (int t = 0; t < 4; ++t) {
      v16h bf = load_b(b0 + t * 16 * DIM, DIM, k0);
      acc[t]     = wmma16(a0, bf, acc[t]);
      acc[4 + t] = wmma16(a1, bf, acc[4 + t]);
    }
  }

  const int half = lane >> 4, l15 = lane & 15;
  #pragma unroll
  for (int t = 0; t < 4; ++t) {
    const int c = cs * 64 + t * 16 + l15;
    const float bv = bias[c];
    #pragma unroll
    for (int rblk = 0; rblk < 2; ++rblk) {
      #pragma unroll
      for (int v = 0; v < 8; ++v) {
        const int r = rb * 32 + rblk * 16 + v + 8 * half;
        out[(size_t)r * DIM + c] = acc[rblk * 4 + t][v] + bv;
      }
    }
  }
}

// ===========================================================================
// Host launcher
// ===========================================================================
extern "C" void kernel_launch(void* const* d_in, const int* in_sizes, int n_in,
                              void* d_out, int out_size, void* d_ws, size_t ws_size,
                              hipStream_t stream) {
  (void)in_sizes; (void)n_in; (void)out_size; (void)ws_size;
  const float* x      = (const float*)d_in[0];  // [8,1024,768]
  const float* w_qkv  = (const float*)d_in[1];  // [768,2304]
  const float* b_qkv  = (const float*)d_in[2];  // [2304]
  const float* w_proj = (const float*)d_in[3];  // [768,768]
  const float* b_proj = (const float*)d_in[4];  // [768]
  float* out = (float*)d_out;

  _Float16* ws = (_Float16*)d_ws;
  _Float16* xh     = ws;                              // 8192*768
  _Float16* wqkvT  = xh     + (size_t)MROWS * DIM;    // 2304*768
  _Float16* wprojT = wqkvT  + (size_t)TDIM  * DIM;    // 768*768
  _Float16* qb     = wprojT + (size_t)DIM   * DIM;    // B*H*N*64
  _Float16* kbuf   = qb     + (size_t)BATCH * HEADS * NSEQ * HD;
  _Float16* vtb    = kbuf   + (size_t)BATCH * HEADS * NSEQ * HD;
  _Float16* wabuf  = vtb    + (size_t)BATCH * HEADS * NSEQ * HD;

  const int nx = MROWS * DIM;
  cvt_f32_f16 <<<(nx + 255) / 256, 256, 0, stream>>>(x, xh, nx);
  trans_f32_f16<<<(DIM * TDIM + 255) / 256, 256, 0, stream>>>(w_qkv, wqkvT, DIM, TDIM);
  trans_f32_f16<<<(DIM * DIM + 255) / 256, 256, 0, stream>>>(w_proj, wprojT, DIM, DIM);

  qkv_gemm <<<(256 * 36) / 4, 128, 0, stream>>>(xh, wqkvT, b_qkv, qb, kbuf, vtb);
  attn_kernel<<<(96 * 64) / 4, 128, 0, stream>>>(qb, kbuf, vtb, wabuf);
  proj_gemm<<<(256 * 12) / 4, 128, 0, stream>>>(wabuf, wprojT, b_proj, out);
}
